// TCRGenAttention_88184268521484
// MI455X (gfx1250) — compile-verified, matrix operand order checked by reading, not run on hardware
//
#include <hip/hip_runtime.h>
#include <hip/hip_bf16.h>

// MI455X / gfx1250, wave32. All matrix math via v_wmma_f32_16x16x32_f16.
// Working set (~110 MB incl. ws) is L2-resident on the 192MB L2 -> matrix-
// engine bound. f16 WMMA everywhere, f32 accumulation + f32 softmax stats.
// K/V tiles in attention are streamed into LDS with the Tensor Data Mover
// (tensor_load_to_lds + s_wait_tensorcnt).

typedef __attribute__((ext_vector_type(16))) _Float16 v16h;
typedef __attribute__((ext_vector_type(8)))  _Float16 v8h;
typedef __attribute__((ext_vector_type(8)))  float    v8f;
typedef __attribute__((ext_vector_type(4)))  unsigned int v4u;
typedef __attribute__((ext_vector_type(8)))  int      v8i;
typedef __attribute__((ext_vector_type(4)))  int      v4i;

#define H_DIM  1024
#define NHEAD  16
#define HD     64
#define ROT    32
#define S_LEN  2048
#define BATCH  2
#define ROWS   (BATCH * S_LEN)   // 4096
#define QKV_N  (3 * H_DIM)       // 3072

#if __has_builtin(__builtin_amdgcn_tensor_load_to_lds)
#define USE_TDM 1
#else
#define USE_TDM 0
#endif

// 16-bit A/B fragment half index j (0..15) -> K index (0..31), ISA 7.12.2.
// Key property: j 0..7 -> K = j + 8*hi (contiguous), j 8..15 -> K = 16+(j-8)+8*hi.
__device__ __forceinline__ int kidx16(int j, int lane) {
  return ((j >> 1) & 3) * 2 + (j & 1) + ((j >> 3) << 4) + ((lane >> 4) << 3);
}

// 8 contiguous source elements -> 8 packed halves (b128 load or v8f load + cvt)
__device__ __forceinline__ v8h cvt8(const _Float16* p) { return *(const v8h*)p; }
__device__ __forceinline__ v8h cvt8(const float* p) {
  v8f f = *(const v8f*)p;
  v8h r;
#pragma unroll
  for (int i = 0; i < 8; ++i) r[i] = (_Float16)f[i];
  return r;
}

union frag16 { v16h v; v8h h[2]; };

// ---- Tensor Data Mover: stage a 32x64 f16 tile (rows x HD) into LDS --------
__device__ __forceinline__ void stage_tile_32x64(const _Float16* __restrict__ gsrc,
                                                 _Float16* __restrict__ ldst,
                                                 int lane) {
#if USE_TDM
  // D# group 0: count=1 | lds_addr | global_addr[56:0] | type=2
  const unsigned long long ga = (unsigned long long)(size_t)gsrc;
  v4u g0;
  g0[0] = 1u;                                   // count = 1 valid descriptor
  g0[1] = (unsigned)(size_t)ldst;               // LDS byte address (flat low 32)
  g0[2] = (unsigned)ga;                         // global_addr[31:0]
  g0[3] = ((unsigned)(ga >> 32) & 0x01FFFFFFu)  // global_addr[56:32]
          | 0x80000000u;                        // type = 2 ("image")
  // D# group 1: data_size=2B; tensor [2048 x 64] elems, dim0_stride=64;
  // tile 32 rows x 64 cols.
  v8i g1;
  g1[0] = 1 << 16;                              // data_size = 1 (2 bytes)
  g1[1] = (HD & 0xFFFF) << 16;                  // tensor_dim0[15:0] @ bits 63:48
  g1[2] = (S_LEN & 0xFFFF) << 16;               // tensor_dim1[15:0] @ bits 95:80 (low half)
  g1[3] = HD << 16;                             // tile_dim0 @ bits 127:112
  g1[4] = 32;                                   // tile_dim1 = 32, tile_dim2 = 0
  g1[5] = HD;                                   // tensor_dim0_stride[31:0]
  g1[6] = 0;
  g1[7] = 0;
  const v4i z4 = {0, 0, 0, 0};
  const v8i z8 = {0, 0, 0, 0, 0, 0, 0, 0};
  // 6-arg form (clang-23 / therock headers): g0, g1, g2, g3, g4, cpol
  __builtin_amdgcn_tensor_load_to_lds(g0, g1, z4, z4, z8, 0);
#else
  // cooperative wave copy, 16B chunks (2048 halves = 32 lanes * 8 chunks)
#pragma unroll
  for (int c = 0; c < 8; ++c)
    *(v8h*)(ldst + (c * 32 + lane) * 8) = *(const v8h*)(gsrc + (c * 32 + lane) * 8);
#endif
}

__device__ __forceinline__ void wait_stage() {
#if USE_TDM
#if __has_builtin(__builtin_amdgcn_s_wait_tensorcnt)
  __builtin_amdgcn_s_wait_tensorcnt(0);
#else
  asm volatile("s_wait_tensorcnt 0x0" ::: "memory");
#endif
  asm volatile("" ::: "memory");
#else
  asm volatile("s_wait_dscnt 0x0" ::: "memory");
#endif
}

// One wave computes one 16x16 C tile of A[MxK] * B[KxN], K-loop step 32.
template <typename TA, typename TB, typename TC>
__device__ __forceinline__ void wmma_gemm_tile(const TA* __restrict__ A,
                                               const TB* __restrict__ B,
                                               TC* __restrict__ C,
                                               int M, int N, int K,
                                               int tm, int tn) {
  const int lane = threadIdx.x & 31;
  const int nrow = lane & 15;
  const int hi   = lane >> 4;
  const size_t arow = (size_t)(tm * 16 + nrow) * K;
  v8f acc = {};
  for (int k0 = 0; k0 < K; k0 += 32) {
    if (k0 + 32 < K)
      __builtin_prefetch(&A[arow + k0 + 32], 0, 1);   // global_prefetch_b8
    frag16 af, bf;
    af.h[0] = cvt8(&A[arow + k0 + hi * 8]);           // K j..j+7  (=j+8hi)
    af.h[1] = cvt8(&A[arow + k0 + 16 + hi * 8]);      // K 16+j+8hi
#pragma unroll
    for (int j = 0; j < 16; ++j) {                    // B is N-strided: scalar
      const int kk = kidx16(j, lane);
      bf.v[j] = (_Float16)B[(size_t)(k0 + kk) * N + tn * 16 + nrow];
    }
    acc = __builtin_amdgcn_wmma_f32_16x16x32_f16(false, af.v, false, bf.v,
                                                 (short)0, acc, false, false);
  }
#pragma unroll
  for (int r = 0; r < 8; ++r)
    C[(size_t)(tm * 16 + r + hi * 8) * N + tn * 16 + nrow] = (TC)acc[r];
}

// ---------------- Kernel 1: QKV projection (fp32 in -> f16 qkv) -------------
__global__ void qkv_gemm_kernel(const float* __restrict__ X,
                                const float* __restrict__ Wqkv,
                                _Float16* __restrict__ QKV) {
  const int w  = threadIdx.x >> 5;
  const int tn = blockIdx.x * 8 + w;   // 192 N-tiles
  const int tm = blockIdx.y;           // 256 M-tiles
  wmma_gemm_tile<float, float, _Float16>(X, Wqkv, QKV, ROWS, QKV_N, H_DIM, tm, tn);
}

// ---------------- Kernel 2: RoPE + head split/scatter -----------------------
__global__ void rope_split_kernel(const _Float16* __restrict__ QKV,
                                  _Float16* __restrict__ Q,
                                  _Float16* __restrict__ Ko,
                                  _Float16* __restrict__ V) {
  const int idx = blockIdx.x * blockDim.x + threadIdx.x;  // ROWS*NHEAD*32
  if (idx >= ROWS * NHEAD * 32) return;
  const int pair = idx & 31;
  const int h    = (idx >> 5) & (NHEAD - 1);
  const int row  = idx >> 9;
  const int b = row >> 11, s = row & (S_LEN - 1);
  const int cbase = (h >> 1) * 384 + (h & 1) * 64;
  const _Float16* src = QKV + (size_t)row * QKV_N;
  const int d = pair * 2;

  float q0 = (float)src[cbase + d],       q1 = (float)src[cbase + d + 1];
  float k0 = (float)src[cbase + 128 + d], k1 = (float)src[cbase + 128 + d + 1];
  float v0 = (float)src[cbase + 256 + d], v1 = (float)src[cbase + 256 + d + 1];

  if (d < ROT) {
    const float freq = __powf(10000.f, -(float)d * (1.f / (float)ROT));
    const float ang  = (float)s * freq;
    const float sn = __sinf(ang), cs = __cosf(ang);
    const float nq0 = q0 * cs - q1 * sn, nq1 = q1 * cs + q0 * sn;
    const float nk0 = k0 * cs - k1 * sn, nk1 = k1 * cs + k0 * sn;
    q0 = nq0; q1 = nq1; k0 = nk0; k1 = nk1;
  }
  const size_t o = (((size_t)(b * NHEAD + h)) * S_LEN + s) * HD + d;
  Q[o]  = (_Float16)q0;  Q[o + 1]  = (_Float16)q1;
  Ko[o] = (_Float16)k0;  Ko[o + 1] = (_Float16)k1;
  V[o]  = (_Float16)v0;  V[o + 1]  = (_Float16)v1;
}

// ---------------- Kernel 3: causal flash attention (WMMA + TDM) -------------
// One wave per (b, h, 16-row q tile); 8 waves/block, each with a private LDS
// region: K tile 32x64 | V tile 32x64 | P tile 16x32 (col-major). No block
// barriers (divergent causal trip counts) -> wave-scoped waits only.
__global__ void flash_attn_kernel(const _Float16* __restrict__ Q,
                                  const _Float16* __restrict__ Kk,
                                  const _Float16* __restrict__ V,
                                  _Float16* __restrict__ O) {
  __shared__ _Float16 lds[8 * 4608];   // 8 waves * (2048 K + 2048 V + 512 P)
  const int lane = threadIdx.x & 31;
  const int w    = threadIdx.x >> 5;
  const int qt   = blockIdx.x * 8 + w;
  const int h    = blockIdx.y;
  const int b    = blockIdx.z;
  const size_t base = ((size_t)(b * NHEAD + h)) * S_LEN * HD;
  const int nrow = lane & 15;
  const int hi   = lane >> 4;
  _Float16* ldsK = lds + w * 4608;
  _Float16* ldsV = ldsK + 2048;
  _Float16* pw   = ldsK + 4096;

  // Q fragments for d=0..31 and d=32..63 (loaded once, b128 chunks)
  frag16 qf0, qf1;
  {
    const _Float16* qr = Q + base + (size_t)(qt * 16 + nrow) * HD;
    qf0.h[0] = cvt8(qr + hi * 8);
    qf0.h[1] = cvt8(qr + 16 + hi * 8);
    qf1.h[0] = cvt8(qr + 32 + hi * 8);
    qf1.h[1] = cvt8(qr + 48 + hi * 8);
  }

  v8f o0 = {}, o1 = {}, o2 = {}, o3 = {};
  float mstat[8], lstat[8];
#pragma unroll
  for (int r = 0; r < 8; ++r) { mstat[r] = -1e30f; lstat[r] = 0.f; }

  const int qhi = qt * 16 + 15;
  for (int k0 = 0; k0 <= qhi; k0 += 32) {
    // ---- stream K and V 32x64 tiles into LDS (TDM) ----
    stage_tile_32x64(Kk + base + (size_t)k0 * HD, ldsK, lane);
    stage_tile_32x64(V  + base + (size_t)k0 * HD, ldsV, lane);
    wait_stage();

    float p0[8], p1[8];
#pragma unroll
    for (int sub = 0; sub < 2; ++sub) {
      // B fragment for scores: lane = key col, halves = d (contiguous rows)
      const _Float16* krow = ldsK + (sub * 16 + nrow) * HD;
      frag16 kf0, kf1;
      kf0.h[0] = *(const v8h*)(krow + hi * 8);
      kf0.h[1] = *(const v8h*)(krow + 16 + hi * 8);
      kf1.h[0] = *(const v8h*)(krow + 32 + hi * 8);
      kf1.h[1] = *(const v8h*)(krow + 48 + hi * 8);
      v8f s = {};
      s = __builtin_amdgcn_wmma_f32_16x16x32_f16(false, qf0.v, false, kf0.v,
                                                 (short)0, s, false, false);
      s = __builtin_amdgcn_wmma_f32_16x16x32_f16(false, qf1.v, false, kf1.v,
                                                 (short)0, s, false, false);
      float* pp = sub ? p1 : p0;
#pragma unroll
      for (int r = 0; r < 8; ++r) {
        const int qrow = qt * 16 + r + hi * 8;
        const int kcol = k0 + sub * 16 + nrow;
        const float sv = s[r] * 0.125f;            // 1/sqrt(64)
        pp[r] = (kcol > qrow) ? -1.0e9f : sv;      // causal mask
      }
    }
    // ---- online softmax: row stats across 16-lane halves ----
    float alpha[8];
#pragma unroll
    for (int r = 0; r < 8; ++r) {
      float rm = fmaxf(p0[r], p1[r]);
#pragma unroll
      for (int off = 1; off < 16; off <<= 1)
        rm = fmaxf(rm, __shfl_xor(rm, off, 32));
      const float mn = fmaxf(mstat[r], rm);
      alpha[r] = __expf(mstat[r] - mn);
      mstat[r] = mn;
      p0[r] = __expf(p0[r] - mn);
      p1[r] = __expf(p1[r] - mn);
      float rs = p0[r] + p1[r];
#pragma unroll
      for (int off = 1; off < 16; off <<= 1)
        rs += __shfl_xor(rs, off, 32);
      lstat[r] = lstat[r] * alpha[r] + rs;
    }
#pragma unroll
    for (int r = 0; r < 8; ++r) {
      o0[r] *= alpha[r]; o1[r] *= alpha[r];
      o2[r] *= alpha[r]; o3[r] *= alpha[r];
    }
    // ---- stage P (16x32) column-major in LDS: one b128 store per column ----
    {
      v8h pk0, pk1;
#pragma unroll
      for (int r = 0; r < 8; ++r) { pk0[r] = (_Float16)p0[r]; pk1[r] = (_Float16)p1[r]; }
      *(v8h*)(pw + (0 * 16 + nrow) * 16 + hi * 8) = pk0;
      *(v8h*)(pw + (1 * 16 + nrow) * 16 + hi * 8) = pk1;
    }
    asm volatile("s_wait_dscnt 0x0" ::: "memory");
    v16h pf;
#pragma unroll
    for (int j = 0; j < 16; ++j)
      pf[j] = pw[kidx16(j, lane) * 16 + nrow];
    asm volatile("s_wait_dscnt 0x0" ::: "memory");
    // ---- P @ V into four 16-wide d tiles (V read from LDS) ----
#pragma unroll
    for (int t = 0; t < 4; ++t) {
      v16h vf;
#pragma unroll
      for (int j = 0; j < 16; ++j)
        vf[j] = ldsV[kidx16(j, lane) * HD + t * 16 + nrow];
      v8f* oacc = (t == 0) ? &o0 : (t == 1) ? &o1 : (t == 2) ? &o2 : &o3;
      *oacc = __builtin_amdgcn_wmma_f32_16x16x32_f16(false, pf, false, vf,
                                                     (short)0, *oacc, false, false);
    }
  }
  // ---- normalize and write attn out in [b][s][h*64+d] layout ----
#pragma unroll
  for (int t = 0; t < 4; ++t) {
    const v8f& oa = (t == 0) ? o0 : (t == 1) ? o1 : (t == 2) ? o2 : o3;
#pragma unroll
    for (int r = 0; r < 8; ++r) {
      const float val = oa[r] / lstat[r];
      O[((size_t)(b * S_LEN + qt * 16 + r + hi * 8)) * H_DIM +
        h * HD + t * 16 + nrow] = (_Float16)val;
    }
  }
}

// ---------------- Kernel 4: output projection (f16 attn -> fp32 out) --------
__global__ void out_gemm_kernel(const _Float16* __restrict__ A,
                                const float* __restrict__ Wout,
                                float* __restrict__ Y) {
  const int w  = threadIdx.x >> 5;
  const int tn = blockIdx.x * 8 + w;   // 64 N-tiles
  const int tm = blockIdx.y;           // 256 M-tiles
  wmma_gemm_tile<_Float16, float, float>(A, Wout, Y, ROWS, H_DIM, H_DIM, tm, tn);
}

extern "C" void kernel_launch(void* const* d_in, const int* in_sizes, int n_in,
                              void* d_out, int out_size, void* d_ws, size_t ws_size,
                              hipStream_t stream) {
  (void)in_sizes; (void)n_in; (void)out_size; (void)ws_size;
  const float* hidden = (const float*)d_in[0];
  const float* w_qkv  = (const float*)d_in[1];
  const float* w_out  = (const float*)d_in[2];
  float* out = (float*)d_out;

  // workspace layout (f16): qkv 24MB | q 8MB | k 8MB | v 8MB | attn 8MB = 56MB
  char* ws = (char*)d_ws;
  _Float16* qkv = (_Float16*)ws;
  _Float16* q   = (_Float16*)(ws + (size_t)ROWS * QKV_N * sizeof(_Float16));
  _Float16* k   = q + (size_t)ROWS * H_DIM;
  _Float16* v   = k + (size_t)ROWS * H_DIM;
  _Float16* att = v + (size_t)ROWS * H_DIM;

  qkv_gemm_kernel<<<dim3(QKV_N / 16 / 8, ROWS / 16), 256, 0, stream>>>(hidden, w_qkv, qkv);
  rope_split_kernel<<<(ROWS * NHEAD * 32) / 256, 256, 0, stream>>>(qkv, q, k, v);
  flash_attn_kernel<<<dim3(S_LEN / 16 / 8, NHEAD, BATCH), 256, 0, stream>>>(q, k, v, att);
  out_gemm_kernel<<<dim3(H_DIM / 16 / 8, ROWS / 16), 256, 0, stream>>>(att, w_out, out);
}
